// MarketBDHPretrain_61555471286948
// MI455X (gfx1250) — matrix operand order repaired and checked by reading, not compiled, not run on hardware
//
#include <hip/hip_runtime.h>
#include <hip/hip_bf16.h>

// ---------------------------------------------------------------------------
// MarketBDHPretrain forward for MI455X (gfx1250, wave32, WMMA bf16/f32-acc).
// Round 3: 64-deep LDS staging via async global->LDS engine, split barriers
// (s_wait_dscnt + s_barrier_signal/-wait) so global prefetches stay in flight
// across the barrier; 8 WMMAs per block sync.
// ---------------------------------------------------------------------------

#define BB       2
#define TT       512
#define DD       256
#define NHH      4
#define NII      4096
#define IN_DIMM  372
#define NLL      6
#define ROWS     (BB * TT)       // 1024 token rows
#define EPSF     1e-5f

#define USE_ASYNC_LDS 1

typedef __attribute__((ext_vector_type(16))) __bf16        v16bf;
typedef __attribute__((ext_vector_type(8)))  float         v8f;
typedef __attribute__((ext_vector_type(8)))  unsigned short ushortx8;

// ---- bf16 <-> f32 (round to nearest even via bit trick) --------------------
__device__ __forceinline__ unsigned short f2bf_bits(float f) {
    unsigned u = __builtin_bit_cast(unsigned, f);
    unsigned r = u + 0x7FFFu + ((u >> 16) & 1u);
    return (unsigned short)(r >> 16);
}
__device__ __forceinline__ float bf2f(unsigned short b) {
    return __builtin_bit_cast(float, ((unsigned)b) << 16);
}

// ---- async global -> LDS copy (16 bytes per enabled lane) ------------------
// Generic pointers to LDS carry the LDS byte offset in their low 32 bits.
__device__ __forceinline__ void lds_cp16(unsigned short* l, const unsigned short* g) {
#if USE_ASYNC_LDS
    unsigned           loff = (unsigned)(size_t)l;
    unsigned long long ga   = (unsigned long long)(size_t)g;
    asm volatile("global_load_async_to_lds_b128 %0, %1, off"
                 :: "v"(loff), "v"(ga) : "memory");
#else
    *(uint4*)l = *(const uint4*)g;   // global_load_b128 + ds_store_b128
#endif
}
__device__ __forceinline__ void async_wait() {
#if USE_ASYNC_LDS
    asm volatile("s_wait_asynccnt 0x0" ::: "memory");
#endif
}
// Split barrier: only DS consistency is required inside the GEMM pipelines
// (async copies are covered by s_wait_asynccnt). Leaves LOADcnt outstanding
// so next-iteration global A/B fragment loads stay in flight across the sync.
__device__ __forceinline__ void block_sync_lds() {
    asm volatile("s_wait_dscnt 0x0\n"
                 "s_barrier_signal -1\n"
                 "s_barrier_wait -1" ::: "memory");
}

// ---- WMMA fragment loaders (layouts from CDNA5 ISA §7.12.2) -----
// A 16x32 bf16, row-major. lanes 0-15: row M=lane, elems 0..7 = K0..7,
// elems 8..15 = K16..23; lanes 16-31: row M=lane-16, K 8..15 / 24..31.
__device__ __forceinline__ v16bf load_frag_a(const unsigned short* A, int lda,
                                             int m0, int k0, int lane) {
    int half = lane >> 4, r = lane & 15;
    const unsigned short* p = A + (size_t)(m0 + r) * lda + (k0 + half * 8);
    union { v16bf v; ushortx8 h[2]; } u;
    u.h[0] = *(const ushortx8*)p;
    u.h[1] = *(const ushortx8*)(p + 16);
    return u.v;
}
// B 32x16 bf16 from N-major storage BT[n][k]. lanes 0-15: col N=lane,
// K 0..7 / 16..23; lanes 16-31: col N=lane-16, K 8..15 / 24..31.
__device__ __forceinline__ v16bf load_frag_b(const unsigned short* BT, int ldb,
                                             int n0, int k0, int lane) {
    int half = lane >> 4, c = lane & 15;
    const unsigned short* p = BT + (size_t)(n0 + c) * ldb + (k0 + half * 8);
    union { v16bf v; ushortx8 h[2]; } u;
    u.h[0] = *(const ushortx8*)p;
    u.h[1] = *(const ushortx8*)(p + 16);
    return u.v;
}
// LDS-tile variants: tile stored as [rows][64] ushorts (K-depth 64), kk = 0|32.
__device__ __forceinline__ v16bf load_frag_a_lds(const unsigned short* buf, int kk, int lane) {
    int half = lane >> 4, r = lane & 15;
    const unsigned short* p = buf + r * 64 + kk + half * 8;
    union { v16bf v; ushortx8 h[2]; } u;
    u.h[0] = *(const ushortx8*)p;
    u.h[1] = *(const ushortx8*)(p + 16);
    return u.v;
}
__device__ __forceinline__ v16bf load_frag_b_lds(const unsigned short* buf, int nloc,
                                                 int kk, int lane) {
    int half = lane >> 4, c = lane & 15;
    const unsigned short* p = buf + (nloc + c) * 64 + kk + half * 8;
    union { v16bf v; ushortx8 h[2]; } u;
    u.h[0] = *(const ushortx8*)p;
    u.h[1] = *(const ushortx8*)(p + 16);
    return u.v;
}

__device__ __forceinline__ v8f wmma_bf16(v16bf a, v16bf b, v8f c) {
    return __builtin_amdgcn_wmma_f32_16x16x32_bf16(false, a, false, b, (short)0, c,
                                                   false, false);
}

// ---- cooperative tile stage: NROWS x 64 ushorts, 16B chunks ----------------
__device__ __forceinline__ void stage_chunk(unsigned short* dst, const unsigned short* src,
                                            int ld, int row0, int k0, int c) {
    int row = c >> 3, koff = (c & 7) * 8;
    lds_cp16(dst + row * 64 + koff, src + (size_t)(row0 + row) * ld + (k0 + koff));
}
template <int NROWS>
__device__ __forceinline__ void stage_tile(unsigned short* dst, const unsigned short* src,
                                           int ld, int row0, int k0, int tid) {
    constexpr int CHUNKS = NROWS * 8;
    if constexpr (CHUNKS == 512) {          // 64-row B tile: 2 copies/thread
        stage_chunk(dst, src, ld, row0, k0, tid);
        stage_chunk(dst, src, ld, row0, k0, tid + 256);
    } else if constexpr (CHUNKS == 256) {   // every thread one copy
        stage_chunk(dst, src, ld, row0, k0, tid);
    } else {                                 // 16-row A tile: tid < 128
        if (tid < CHUNKS) stage_chunk(dst, src, ld, row0, k0, tid);
    }
}

// ===========================================================================
// Block GEMM, B staged in LDS (waves split M; B tile shared by all 8 waves).
// Wave strip: 16 x (16*NT).  LDS: 2 x (16*NT x 64) ushorts, double buffered.
// K advances 64 per barrier -> 2*NT WMMAs per sync.
// ===========================================================================
template <int NT>
__device__ __forceinline__ void gemm_block_bstaged(
    const unsigned short* A, int lda, int m_wave,
    const unsigned short* BT, int ldb, int n0,
    int K, int tid, int lane, unsigned short* sB, v8f acc[NT]) {
    const int TILE = 16 * NT * 64;
    stage_tile<16 * NT>(sB, BT, ldb, n0, 0, tid);
    v16bf a0 = load_frag_a(A, lda, m_wave, 0, lane);
    v16bf a1 = load_frag_a(A, lda, m_wave, 32, lane);
    async_wait();
    block_sync_lds();
    int cur = 0;
    for (int k0 = 0; k0 < K; k0 += 64) {
        int nxt = cur ^ 1;
        bool more = (k0 + 64) < K;
        if (more) stage_tile<16 * NT>(sB + nxt * TILE, BT, ldb, n0, k0 + 64, tid);
        v16bf a2 = a0, a3 = a1;
        if (more) {
            a2 = load_frag_a(A, lda, m_wave, k0 + 64, lane);
            a3 = load_frag_a(A, lda, m_wave, k0 + 96, lane);
        }
#pragma unroll
        for (int t = 0; t < NT; ++t)
            acc[t] = wmma_bf16(a0, load_frag_b_lds(sB + cur * TILE, 16 * t, 0, lane), acc[t]);
#pragma unroll
        for (int t = 0; t < NT; ++t)
            acc[t] = wmma_bf16(a1, load_frag_b_lds(sB + cur * TILE, 16 * t, 32, lane), acc[t]);
        a0 = a2; a1 = a3;
        async_wait();
        block_sync_lds();
        cur = nxt;
    }
}

// ===========================================================================
// Block GEMM, A staged in LDS (waves split N over 16 shared rows).
// Wave strip: 16 x (16*NT).  LDS: 2 x (16 x 64) ushorts, double buffered.
// B fragments double-buffered in registers.
// ===========================================================================
template <int NT>
__device__ __forceinline__ void gemm_block_astaged(
    const unsigned short* A, int lda, int m0,
    const unsigned short* BT, int ldb, int n_wave,
    int K, int tid, int lane, unsigned short* sA, v8f acc[NT]) {
    const int TILE = 16 * 64;
    stage_tile<16>(sA, A, lda, m0, 0, tid);
    v16bf b0[NT], b1[NT];
#pragma unroll
    for (int t = 0; t < NT; ++t) {
        b0[t] = load_frag_b(BT, ldb, n_wave + 16 * t, 0, lane);
        b1[t] = load_frag_b(BT, ldb, n_wave + 16 * t, 32, lane);
    }
    async_wait();
    block_sync_lds();
    int cur = 0;
    for (int k0 = 0; k0 < K; k0 += 64) {
        int nxt = cur ^ 1;
        bool more = (k0 + 64) < K;
        if (more) stage_tile<16>(sA + nxt * TILE, A, lda, m0, k0 + 64, tid);
        v16bf c0[NT], c1[NT];
#pragma unroll
        for (int t = 0; t < NT; ++t) { c0[t] = b0[t]; c1[t] = b1[t]; }
        if (more) {
#pragma unroll
            for (int t = 0; t < NT; ++t) {
                c0[t] = load_frag_b(BT, ldb, n_wave + 16 * t, k0 + 64, lane);
                c1[t] = load_frag_b(BT, ldb, n_wave + 16 * t, k0 + 96, lane);
            }
        }
        v16bf a0 = load_frag_a_lds(sA + cur * TILE, 0, lane);
        v16bf a1 = load_frag_a_lds(sA + cur * TILE, 32, lane);
#pragma unroll
        for (int t = 0; t < NT; ++t) acc[t] = wmma_bf16(a0, b0[t], acc[t]);
#pragma unroll
        for (int t = 0; t < NT; ++t) acc[t] = wmma_bf16(a1, b1[t], acc[t]);
#pragma unroll
        for (int t = 0; t < NT; ++t) { b0[t] = c0[t]; b1[t] = c1[t]; }
        async_wait();
        block_sync_lds();
        cur = nxt;
    }
}

// C/D 16x16 f32 tile: lane 0-15 -> N=lane, VGPR r -> M=r; lanes 16-31 -> M=r+8
__device__ __forceinline__ void store_tile_bf16(unsigned short* C, int ldc,
                                                int m0, int n0, int lane, const v8f acc) {
    int n = lane & 15, mo = (lane >> 4) * 8;
#pragma unroll
    for (int r = 0; r < 8; ++r)
        C[(size_t)(m0 + mo + r) * ldc + (n0 + n)] = f2bf_bits(acc[r]);
}
__device__ __forceinline__ void store_tile_relu_bf16(unsigned short* C, int ldc,
                                                     int m0, int n0, int lane, const v8f acc) {
    int n = lane & 15, mo = (lane >> 4) * 8;
#pragma unroll
    for (int r = 0; r < 8; ++r) {
        float v = acc[r]; v = v > 0.f ? v : 0.f;
        C[(size_t)(m0 + mo + r) * ldc + (n0 + n)] = f2bf_bits(v);
    }
}

// ===========================================================================
// Weight prep: one-time (per call) f32 -> bf16 transposed copies.
// ===========================================================================
__global__ void k_prep_encT(const float* __restrict__ enc, unsigned short* __restrict__ encT) {
    // enc (NH, D, NI) -> encT[h][n][d]
    size_t i = (size_t)blockIdx.x * blockDim.x + threadIdx.x;
    if (i >= (size_t)NHH * NII * DD) return;
    int d = (int)(i % DD); size_t r = i / DD;
    int n = (int)(r % NII); int h = (int)(r / NII);
    encT[i] = f2bf_bits(enc[((size_t)h * DD + d) * NII + n]);
}
__global__ void k_prep_decT(const float* __restrict__ dec, unsigned short* __restrict__ decT) {
    // dec (NH*NI, D) -> decT[h][d][n]
    size_t i = (size_t)blockIdx.x * blockDim.x + threadIdx.x;
    if (i >= (size_t)NHH * DD * NII) return;
    int n = (int)(i % NII); size_t r = i / NII;
    int d = (int)(r % DD);  int h = (int)(r / DD);
    decT[i] = f2bf_bits(dec[((size_t)h * NII + n) * DD + d]);
}

// ===========================================================================
// Input projection: tmp[row][d] = x_flat[row] . in_W[:,d] + in_b[d]   (small)
// ===========================================================================
__global__ __launch_bounds__(256) void k_inproj(const float* __restrict__ x,
                                                const float* __restrict__ W,
                                                const float* __restrict__ bvec,
                                                float* __restrict__ tmp) {
    int row = blockIdx.x, d = threadIdx.x;
    const float* xr = x + (size_t)row * IN_DIMM;
    float acc = bvec[d];
    for (int k = 0; k < IN_DIMM; ++k) acc = fmaf(xr[k], W[(size_t)k * DD + d], acc);
    tmp[(size_t)row * DD + d] = acc;
}

// LN one row per block; emit h in all three forms (f32, bf16, bf16-transposed)
__global__ __launch_bounds__(256) void k_ln_h(const float* __restrict__ tmp,
                                              float* __restrict__ h32,
                                              unsigned short* __restrict__ hbf,
                                              unsigned short* __restrict__ hT) {
    __shared__ float s1[256], s2[256];
    int row = blockIdx.x, tid = threadIdx.x;
    float v = tmp[(size_t)row * DD + tid];
    s1[tid] = v; s2[tid] = v * v; __syncthreads();
    for (int off = 128; off > 0; off >>= 1) {
        if (tid < off) { s1[tid] += s1[tid + off]; s2[tid] += s2[tid + off]; }
        __syncthreads();
    }
    float m = s1[0] * (1.f / DD);
    float var = s2[0] * (1.f / DD) - m * m;
    float o = (v - m) * rsqrtf(var + EPSF);
    h32[(size_t)row * DD + tid] = o;
    unsigned short bbits = f2bf_bits(o);
    hbf[(size_t)row * DD + tid] = bbits;
    int b = row >> 9, t = row & (TT - 1);
    hT[((size_t)b * DD + tid) * TT + t] = bbits;
}

// ===========================================================================
// (1) h_sparse = relu(h @ encoder[h])    per head: [1024x256]x[256x4096]
// ===========================================================================
__global__ __launch_bounds__(256) void k_hsparse(const unsigned short* __restrict__ hbf,
                                                 const unsigned short* __restrict__ encT,
                                                 unsigned short* __restrict__ hsp) {
    __shared__ __align__(16) unsigned short sB[2 * 64 * 64];
    int tid = threadIdx.x, wave = tid >> 5, lane = tid & 31;
    int h  = blockIdx.z;
    int m0 = blockIdx.y * 128 + wave * 16;
    int n0 = blockIdx.x * 64;
    const unsigned short* BT = encT + (size_t)h * NII * DD;  // [NI][D]
    v8f acc[4] = {};
    gemm_block_bstaged<4>(hbf, DD, m0, BT, DD, n0, DD, tid, lane, sB, acc);
    unsigned short* O = hsp + (size_t)h * ROWS * NII;
#pragma unroll
    for (int t = 0; t < 4; ++t) store_tile_relu_bf16(O, NII, m0, n0 + 16 * t, lane, acc[t]);
}

// ===========================================================================
// RoPE (elementwise on pairs): qr = rope(h_sparse)
// ===========================================================================
__global__ __launch_bounds__(256) void k_rope(const unsigned short* __restrict__ hsp,
                                              unsigned short* __restrict__ qr) {
    size_t i = (size_t)blockIdx.x * blockDim.x + threadIdx.x;   // pair index
    const size_t npairs = (size_t)NHH * ROWS * (NII / 2);
    if (i >= npairs) return;
    int pair = (int)(i % (NII / 2));
    size_t row = i / (NII / 2);                 // h*ROWS + b*T + t
    int t = (int)(row % TT);
    // freq = theta^(-2i/NI) = 2^(-16*2i/4096) = exp2(-i/128)
    float freq = exp2f(-(float)pair * (1.0f / 128.0f));
    float s, c; __sincosf((float)t * freq, &s, &c);
    unsigned pb = *(const unsigned*)(hsp + row * NII + 2 * pair);
    float x1 = bf2f((unsigned short)(pb & 0xFFFFu));
    float x2 = bf2f((unsigned short)(pb >> 16));
    unsigned ob = (unsigned)f2bf_bits(x1 * c - x2 * s) |
                  ((unsigned)f2bf_bits(x1 * s + x2 * c) << 16);
    *(unsigned*)(qr + row * NII + 2 * pair) = ob;
}

// ===========================================================================
// (2) scores = qr @ qr^T  per (b,h): [512x4096]x[4096x512]  (the big one)
// ===========================================================================
__global__ __launch_bounds__(256) void k_scores(const unsigned short* __restrict__ qr,
                                                unsigned short* __restrict__ sc) {
    __shared__ __align__(16) unsigned short sB[2 * 64 * 64];
    int tid = threadIdx.x, wave = tid >> 5, lane = tid & 31;
    int z = blockIdx.z, b = z / NHH, h = z % NHH;
    const unsigned short* Q = qr + ((size_t)h * ROWS + (size_t)b * TT) * NII;
    int m0 = blockIdx.y * 128 + wave * 16;   // t
    int n0 = blockIdx.x * 64;                // s
    v8f acc[4] = {};
    gemm_block_bstaged<4>(Q, NII, m0, Q, NII, n0, NII, tid, lane, sB, acc);
    unsigned short* O = sc + (size_t)(b * NHH + h) * TT * TT;
#pragma unroll
    for (int t = 0; t < 4; ++t) store_tile_bf16(O, TT, m0, n0 + 16 * t, lane, acc[t]);
}

// ===========================================================================
// (3) yKV = scores @ h, then LN over d.  per (b,h): [512x512]x[512x256]
//     block = 16 rows x 256 cols (A staged in LDS, waves split N) -> LN in LDS
// ===========================================================================
__global__ __launch_bounds__(256) void k_ykv_ln(const unsigned short* __restrict__ sc,
                                                const unsigned short* __restrict__ hT,
                                                unsigned short* __restrict__ ykv) {
    __shared__ __align__(16) unsigned short sA[2 * 16 * 64];
    __shared__ float sC[16][DD + 1];
    int tid = threadIdx.x, wave = tid >> 5, lane = tid & 31;
    int z = blockIdx.z, b = z / NHH, h = z % NHH;
    int m0 = blockIdx.x * 16;
    const unsigned short* A  = sc + (size_t)(b * NHH + h) * TT * TT; // [T][T]
    const unsigned short* BT = hT + (size_t)b * DD * TT;             // [D][T]
    int n0 = wave * 32;
    v8f acc[2] = {};
    gemm_block_astaged<2>(A, TT, m0, BT, TT, n0, TT, tid, lane, sA, acc);
    int n = lane & 15, mo = (lane >> 4) * 8;
#pragma unroll
    for (int t = 0; t < 2; ++t)
#pragma unroll
        for (int r = 0; r < 8; ++r) sC[mo + r][n0 + 16 * t + n] = acc[t][r];
    __syncthreads();
    if (tid < 16) {
        float m = 0.f;
        for (int c = 0; c < DD; ++c) m += sC[tid][c];
        m *= (1.f / DD);
        float v = 0.f;
        for (int c = 0; c < DD; ++c) { float d = sC[tid][c] - m; v += d * d; }
        float rs = rsqrtf(v * (1.f / DD) + EPSF);
        unsigned short* o = ykv + ((size_t)(b * NHH + h) * TT + m0 + tid) * DD;
        for (int c = 0; c < DD; ++c) o[c] = f2bf_bits((sC[tid][c] - m) * rs);
    }
}

// ===========================================================================
// (4) y_sparse = relu(yKV_ln @ encoder_v[h]); xy = h_sparse * y_sparse
// ===========================================================================
__global__ __launch_bounds__(256) void k_ysparse_xy(const unsigned short* __restrict__ ykv,
                                                    const unsigned short* __restrict__ encvT,
                                                    const unsigned short* __restrict__ hsp,
                                                    unsigned short* __restrict__ xy) {
    __shared__ __align__(16) unsigned short sB[2 * 64 * 64];
    int tid = threadIdx.x, wave = tid >> 5, lane = tid & 31;
    int z = blockIdx.z, b = z / NHH, h = z % NHH;
    int m0 = blockIdx.y * 128 + wave * 16;   // t
    int n0 = blockIdx.x * 64;                // n
    const unsigned short* A  = ykv + (size_t)(b * NHH + h) * TT * DD; // [T][D]
    const unsigned short* BT = encvT + (size_t)h * NII * DD;          // [NI][D]
    v8f acc[4] = {};
    gemm_block_bstaged<4>(A, DD, m0, BT, DD, n0, DD, tid, lane, sB, acc);
    const unsigned short* HS = hsp + ((size_t)h * ROWS + (size_t)b * TT) * NII;
    unsigned short*       O  = xy  + ((size_t)h * ROWS + (size_t)b * TT) * NII;
    int n = lane & 15, mo = (lane >> 4) * 8;
#pragma unroll
    for (int t = 0; t < 4; ++t)
#pragma unroll
        for (int r = 0; r < 8; ++r) {
            size_t idx = (size_t)(m0 + mo + r) * NII + (n0 + 16 * t + n);
            float v = acc[t][r]; v = v > 0.f ? v : 0.f;
            O[idx] = f2bf_bits(v * bf2f(HS[idx]));
        }
}

// ===========================================================================
// (5) yMLP = sum_h xy[h] @ dec[h];  h = LN(h_res + LN(yMLP))
// ===========================================================================
__global__ __launch_bounds__(256) void k_ymlp_ln(const unsigned short* __restrict__ xy,
                                                 const unsigned short* __restrict__ decT,
                                                 float* __restrict__ h32,
                                                 unsigned short* __restrict__ hbf,
                                                 unsigned short* __restrict__ hT) {
    __shared__ __align__(16) unsigned short sA[2 * 16 * 64];
    __shared__ float sC[16][DD + 1];
    int tid = threadIdx.x, wave = tid >> 5, lane = tid & 31;
    int m0 = blockIdx.x * 16;                // token-row tile
    int n0 = wave * 32;
    v8f acc[2] = {};
    for (int h = 0; h < NHH; ++h) {
        const unsigned short* A  = xy   + (size_t)h * ROWS * NII;  // [1024][NI]
        const unsigned short* BT = decT + (size_t)h * DD * NII;    // [D][NI]
        gemm_block_astaged<2>(A, NII, m0, BT, NII, n0, NII, tid, lane, sA, acc);
    }
    int n = lane & 15, mo = (lane >> 4) * 8;
#pragma unroll
    for (int t = 0; t < 2; ++t)
#pragma unroll
        for (int r = 0; r < 8; ++r) sC[mo + r][n0 + 16 * t + n] = acc[t][r];
    __syncthreads();
    if (tid < 16) {
        int row = m0 + tid;
        float m1 = 0.f;
        for (int c = 0; c < DD; ++c) m1 += sC[tid][c];
        m1 *= (1.f / DD);
        float v1 = 0.f;
        for (int c = 0; c < DD; ++c) { float d = sC[tid][c] - m1; v1 += d * d; }
        float r1 = rsqrtf(v1 * (1.f / DD) + EPSF);
        for (int c = 0; c < DD; ++c)
            sC[tid][c] = h32[(size_t)row * DD + c] + (sC[tid][c] - m1) * r1;
        float m2 = 0.f;
        for (int c = 0; c < DD; ++c) m2 += sC[tid][c];
        m2 *= (1.f / DD);
        float v2 = 0.f;
        for (int c = 0; c < DD; ++c) { float d = sC[tid][c] - m2; v2 += d * d; }
        float r2 = rsqrtf(v2 * (1.f / DD) + EPSF);
        int b = row >> 9, t = row & (TT - 1);
        for (int c = 0; c < DD; ++c) {
            float o = (sC[tid][c] - m2) * r2;
            h32[(size_t)row * DD + c] = o;
            unsigned short bbits = f2bf_bits(o);
            hbf[(size_t)row * DD + c] = bbits;
            hT[((size_t)b * DD + c) * TT + t] = bbits;
        }
    }
}

// ===========================================================================
// Head: pred[row][j] = h . head_W[:,j] + head_b[j]   (small)
// ===========================================================================
__global__ __launch_bounds__(384) void k_head(const float* __restrict__ h32,
                                              const float* __restrict__ W,
                                              const float* __restrict__ bvec,
                                              float* __restrict__ out) {
    int row = blockIdx.x, j = threadIdx.x;
    if (j >= IN_DIMM) return;
    float acc = bvec[j];
    const float* hr = h32 + (size_t)row * DD;
    for (int k = 0; k < DD; ++k) acc = fmaf(hr[k], W[(size_t)k * IN_DIMM + j], acc);
    out[(size_t)row * IN_DIMM + j] = acc;
}

// ===========================================================================
extern "C" void kernel_launch(void* const* d_in, const int* in_sizes, int n_in,
                              void* d_out, int out_size, void* d_ws, size_t ws_size,
                              hipStream_t stream) {
    const float* x      = (const float*)d_in[0];
    const float* in_W   = (const float*)d_in[1];
    const float* in_b   = (const float*)d_in[2];
    const float* enc    = (const float*)d_in[3];
    const float* encv   = (const float*)d_in[4];
    const float* dec    = (const float*)d_in[5];
    const float* head_W = (const float*)d_in[6];
    const float* head_b = (const float*)d_in[7];
    float* out = (float*)d_out;

    // ---- workspace carve-up (bytes, 256-aligned) ---------------------------
    char* ws = (char*)d_ws;
    size_t off = 0;
    auto carve = [&](size_t bytes) { char* p = ws + off; off = (off + bytes + 255) & ~(size_t)255; return p; };
    unsigned short* encT  = (unsigned short*)carve((size_t)NHH * NII * DD * 2); // 8MB
    unsigned short* encvT = (unsigned short*)carve((size_t)NHH * NII * DD * 2); // 8MB
    unsigned short* decT  = (unsigned short*)carve((size_t)NHH * DD * NII * 2); // 8MB
    float*          h32   = (float*)carve((size_t)ROWS * DD * 4);               // 1MB
    unsigned short* hbf   = (unsigned short*)carve((size_t)ROWS * DD * 2);
    unsigned short* hT    = (unsigned short*)carve((size_t)BB * DD * TT * 2);
    float*          tmp0  = (float*)carve((size_t)ROWS * DD * 4);
    unsigned short* hsp   = (unsigned short*)carve((size_t)NHH * ROWS * NII * 2); // 33.5MB
    unsigned short* qrxy  = (unsigned short*)carve((size_t)NHH * ROWS * NII * 2); // 33.5MB (qr, reused as xy)
    unsigned short* sc    = (unsigned short*)carve((size_t)BB * NHH * TT * TT * 2); // 4.2MB
    unsigned short* ykv   = (unsigned short*)carve((size_t)NHH * ROWS * DD * 2);   // 2MB
    (void)ws_size; (void)in_sizes; (void)n_in; (void)out_size;

    // ---- weight prep -------------------------------------------------------
    {
        size_t n = (size_t)NHH * NII * DD;
        int blocks = (int)((n + 255) / 256);
        k_prep_encT<<<blocks, 256, 0, stream>>>(enc,  encT);
        k_prep_encT<<<blocks, 256, 0, stream>>>(encv, encvT);
        k_prep_decT<<<blocks, 256, 0, stream>>>(dec,  decT);
    }

    // ---- input projection + LN --------------------------------------------
    k_inproj<<<ROWS, DD, 0, stream>>>(x, in_W, in_b, tmp0);
    k_ln_h<<<ROWS, DD, 0, stream>>>(tmp0, h32, hbf, hT);

    // ---- 6 layers ----------------------------------------------------------
    for (int l = 0; l < NLL; ++l) {
        k_hsparse<<<dim3(NII / 64, ROWS / 128, NHH), 256, 0, stream>>>(hbf, encT, hsp);
        {
            size_t npairs = (size_t)NHH * ROWS * (NII / 2);
            k_rope<<<(int)((npairs + 255) / 256), 256, 0, stream>>>(hsp, qrxy);
        }
        k_scores<<<dim3(TT / 64, TT / 128, BB * NHH), 256, 0, stream>>>(qrxy, sc);
        k_ykv_ln<<<dim3(TT / 16, 1, BB * NHH), 256, 0, stream>>>(sc, hT, ykv);
        k_ysparse_xy<<<dim3(NII / 64, TT / 128, BB * NHH), 256, 0, stream>>>(ykv, encvT, hsp, qrxy);
        k_ymlp_ln<<<dim3(ROWS / 16, 1, 1), 256, 0, stream>>>(qrxy, decT, h32, hbf, hT);
    }

    // ---- head --------------------------------------------------------------
    k_head<<<ROWS, 384, 0, stream>>>(h32, head_W, head_b, out);
}
